// NaViT_10926396801224
// MI455X (gfx1250) — matrix operand order repaired and checked by reading, not compiled, not run on hardware
//
#include <hip/hip_runtime.h>
#include <hip/hip_bf16.h>
#include <math.h>

typedef __bf16 bf16_t;
typedef bf16_t v16bf __attribute__((ext_vector_type(16)));
typedef float  v8f   __attribute__((ext_vector_type(8)));

struct alignas(16) bfx8 { bf16_t d[8]; };
union V16u { v16bf v; bfx8 h[2]; };

#define BB      2
#define LL      2048
#define PDIM    768
#define DMODEL  512
#define NH      8
#define DHEAD   64
#define NDEPTH  2
#define DMLP    2048
#define NIMG_   4
#define NCLS_   1000

__device__ __forceinline__ v8f wmma_bf16(v16bf a, v16bf b, v8f c) {
  return __builtin_amdgcn_wmma_f32_16x16x32_bf16(false, a, false, b, (short)0, c, false, false);
}

// ---------------------------------------------------------------------------
// Generic WMMA GEMM: C[M,N] = A_bf16[M,K] * Bt_bf16[N,K]^T (+bias)(+gelu)(+res)
// Block = 128 threads = 4 waves; wave owns 16 rows, block owns 64x64 tile.
// ---------------------------------------------------------------------------
__global__ void __launch_bounds__(128) gemm_bf16_kernel(
    const bf16_t* __restrict__ A, const bf16_t* __restrict__ Bt,
    const float* __restrict__ bias, const float* __restrict__ residual,
    float* __restrict__ outF, bf16_t* __restrict__ outB,
    int M, int N, int K, int do_gelu)
{
  const int lane = threadIdx.x & 31;
  const int wv   = threadIdx.x >> 5;
  const int half = lane >> 4;
  const int l15  = lane & 15;
  const int rBase = blockIdx.y * 64 + wv * 16;
  const int cBase = blockIdx.x * 64;

  v8f c[4] = {};
  const bf16_t* arow = A + (size_t)(rBase + l15) * K;
  for (int k0 = 0; k0 < K; k0 += 32) {
    V16u ua;
    ua.h[0] = *(const bfx8*)(arow + k0 + 16 * half);
    ua.h[1] = *(const bfx8*)(arow + k0 + 8 + 16 * half);
#pragma unroll
    for (int t = 0; t < 4; t++) {
      const bf16_t* bcol = Bt + (size_t)(cBase + t * 16 + l15) * K + k0 + 16 * half;
      v16bf bv = *(const v16bf*)bcol;
      c[t] = wmma_bf16(ua.v, bv, c[t]);
    }
  }
#pragma unroll
  for (int t = 0; t < 4; t++) {
    int col = cBase + t * 16 + l15;
    float bv = bias ? bias[col] : 0.0f;
#pragma unroll
    for (int v = 0; v < 8; v++) {
      int row = rBase + 8 * half + v;
      float x = c[t][v] + bv;
      if (do_gelu) x = 0.5f * x * (1.0f + erff(x * 0.70710678118654752f));
      size_t idx = (size_t)row * N + col;
      if (residual) x += residual[idx];
      if (outF) outF[idx] = x;
      if (outB) outB[idx] = (bf16_t)x;
    }
  }
}

// ---------------------------------------------------------------------------
// Weight convert + transpose: W[K,N] f32 -> Wt[N,K] bf16
// ---------------------------------------------------------------------------
__global__ void convT_kernel(const float* __restrict__ src, bf16_t* __restrict__ dst,
                             int K, int N) {
  int idx = blockIdx.x * 256 + threadIdx.x;
  if (idx >= K * N) return;
  int k = idx / N, n = idx - k * N;
  dst[(size_t)n * K + k] = (bf16_t)src[idx];
}

// ---------------------------------------------------------------------------
// LayerNorm (bias-free): out = (x-mu)/sqrt(var+1e-5)*g ; f32 and/or bf16 out
// one row per block, 256 threads, D <= 768
// ---------------------------------------------------------------------------
__global__ void __launch_bounds__(256) ln_kernel(
    const float* __restrict__ X, const float* __restrict__ g,
    float* __restrict__ outF, bf16_t* __restrict__ outB, int D)
{
  const int row = blockIdx.x;
  const float* x = X + (size_t)row * D;
  float vals[3];
  float s = 0.f, ss = 0.f;
  int nv = 0;
  for (int i = threadIdx.x; i < D; i += 256) {
    float v = x[i];
    vals[nv++] = v;
    s += v; ss += v * v;
  }
  for (int d = 1; d < 32; d <<= 1) { s += __shfl_xor(s, d, 32); ss += __shfl_xor(ss, d, 32); }
  __shared__ float sh1[8], sh2[8], tot[2];
  int wv = threadIdx.x >> 5, lane = threadIdx.x & 31;
  if (lane == 0) { sh1[wv] = s; sh2[wv] = ss; }
  __syncthreads();
  if (threadIdx.x == 0) {
    float a = 0.f, b = 0.f;
    for (int i = 0; i < 8; i++) { a += sh1[i]; b += sh2[i]; }
    tot[0] = a; tot[1] = b;
  }
  __syncthreads();
  float mean = tot[0] / (float)D;
  float var  = tot[1] / (float)D - mean * mean;
  float rstd = rsqrtf(var + 1e-5f);
  nv = 0;
  for (int i = threadIdx.x; i < D; i += 256) {
    float y = (vals[nv++] - mean) * rstd * g[i];
    if (outF) outF[(size_t)row * D + i] = y;
    if (outB) outB[(size_t)row * D + i] = (bf16_t)y;
  }
}

// ---------------------------------------------------------------------------
// Patch-embed post: X = LN(T)*g + pos_h_emb[ph] + pos_w_emb[pw]   (D=512)
// ---------------------------------------------------------------------------
__global__ void __launch_bounds__(256) ln_pos_kernel(
    const float* __restrict__ T, const float* __restrict__ g,
    const float* __restrict__ phe, const float* __restrict__ pwe,
    const int* __restrict__ ph, const int* __restrict__ pw,
    float* __restrict__ X)
{
  const int row = blockIdx.x;
  const float* x = T + (size_t)row * DMODEL;
  int t = threadIdx.x;
  float v0 = x[t], v1 = x[t + 256];
  float s = v0 + v1, ss = v0 * v0 + v1 * v1;
  for (int d = 1; d < 32; d <<= 1) { s += __shfl_xor(s, d, 32); ss += __shfl_xor(ss, d, 32); }
  __shared__ float sh1[8], sh2[8], tot[2];
  int wv = t >> 5, lane = t & 31;
  if (lane == 0) { sh1[wv] = s; sh2[wv] = ss; }
  __syncthreads();
  if (t == 0) {
    float a = 0.f, b = 0.f;
    for (int i = 0; i < 8; i++) { a += sh1[i]; b += sh2[i]; }
    tot[0] = a; tot[1] = b;
  }
  __syncthreads();
  float mean = tot[0] * (1.0f / DMODEL);
  float var  = tot[1] * (1.0f / DMODEL) - mean * mean;
  float rstd = rsqrtf(var + 1e-5f);
  int ih = ph[row], iw = pw[row];
  float y0 = (v0 - mean) * rstd * g[t] + phe[(size_t)ih * DMODEL + t] + pwe[(size_t)iw * DMODEL + t];
  float y1 = (v1 - mean) * rstd * g[t + 256] + phe[(size_t)ih * DMODEL + t + 256] + pwe[(size_t)iw * DMODEL + t + 256];
  X[(size_t)row * DMODEL + t] = y0;
  X[(size_t)row * DMODEL + t + 256] = y1;
}

// ---------------------------------------------------------------------------
// RMS-norm Q/K per head (x/max(|x|,1e-12)*8*g), pack bf16; also V -> Vt[b,h,dh,L]
// one block per (b,n), wave = head, lane handles dh = 2*lane, 2*lane+1
// ---------------------------------------------------------------------------
__global__ void __launch_bounds__(256) rms_prep_kernel(
    const float* __restrict__ Qf, const float* __restrict__ KVf,
    const float* __restrict__ qg, const float* __restrict__ kg,
    bf16_t* __restrict__ Qb, bf16_t* __restrict__ Kb, bf16_t* __restrict__ Vt)
{
  int bn = blockIdx.x;
  int b = bn >> 11, n = bn & (LL - 1);
  int h = threadIdx.x >> 5, lane = threadIdx.x & 31;
  int d0 = lane * 2;
  size_t rq = (size_t)bn * DMODEL + h * DHEAD + d0;
  size_t rk = (size_t)bn * (2 * DMODEL) + h * DHEAD + d0;
  float q0 = Qf[rq], q1 = Qf[rq + 1];
  float k0 = KVf[rk], k1 = KVf[rk + 1];
  float ssq = q0 * q0 + q1 * q1;
  float ssk = k0 * k0 + k1 * k1;
  for (int d = 1; d < 32; d <<= 1) { ssq += __shfl_xor(ssq, d, 32); ssk += __shfl_xor(ssk, d, 32); }
  float sq = 8.0f / fmaxf(sqrtf(ssq), 1e-12f);
  float sk = 8.0f / fmaxf(sqrtf(ssk), 1e-12f);
  size_t ob = (((size_t)(b * NH + h)) * LL + n) * DHEAD + d0;
  Qb[ob]     = (bf16_t)(q0 * sq * qg[h * DHEAD + d0]);
  Qb[ob + 1] = (bf16_t)(q1 * sq * qg[h * DHEAD + d0 + 1]);
  Kb[ob]     = (bf16_t)(k0 * sk * kg[h * DHEAD + d0]);
  Kb[ob + 1] = (bf16_t)(k1 * sk * kg[h * DHEAD + d0 + 1]);
  float v0 = KVf[rk + DMODEL], v1 = KVf[rk + DMODEL + 1];
  size_t vb = ((size_t)(b * NH + h) * DHEAD + d0) * LL + n;
  Vt[vb]      = (bf16_t)v0;
  Vt[vb + LL] = (bf16_t)v1;
}

// ---------------------------------------------------------------------------
// Flash attention with block-diagonal mask. Wave handles 16 query rows,
// iterates key blocks of 32. Output bf16 [b, n, h*64+dh].
// ---------------------------------------------------------------------------
__global__ void __launch_bounds__(128) flash_attn_kernel(
    const bf16_t* __restrict__ Qb, const bf16_t* __restrict__ Kb,
    const bf16_t* __restrict__ Vt, const int* __restrict__ ids,
    bf16_t* __restrict__ AOb)
{
  __shared__ bf16_t P[4][16][32];
  const int lane = threadIdx.x & 31, wv = threadIdx.x >> 5;
  const int half = lane >> 4, l15 = lane & 15;
  const int tile = blockIdx.x * 4 + wv;
  const int q16 = tile & (LL / 16 - 1);
  const int bh = tile >> 7;
  const int h = bh & (NH - 1), b = bh >> 3;
  const size_t bhL = (size_t)bh * LL;

  V16u aq0, aq1;
  {
    const bf16_t* qrow = Qb + (bhL + q16 * 16 + l15) * DHEAD;
    aq0.h[0] = *(const bfx8*)(qrow + 16 * half);
    aq0.h[1] = *(const bfx8*)(qrow + 8 + 16 * half);
    aq1.h[0] = *(const bfx8*)(qrow + 32 + 16 * half);
    aq1.h[1] = *(const bfx8*)(qrow + 40 + 16 * half);
  }
  int qid[8];
#pragma unroll
  for (int v = 0; v < 8; v++) qid[v] = ids[b * LL + q16 * 16 + 8 * half + v];

  float mstat[8], lstat[8];
#pragma unroll
  for (int v = 0; v < 8; v++) { mstat[v] = -__builtin_inff(); lstat[v] = 0.f; }
  v8f o[4] = {};

  const bf16_t* Kp = Kb + bhL * DHEAD;
  const bf16_t* Vp = Vt + bhL * DHEAD;   // == bh*DHEAD*LL

  for (int j0 = 0; j0 < LL; j0 += 32) {
    v8f s[2];
#pragma unroll
    for (int jj = 0; jj < 2; jj++) {
      const bf16_t* krow = Kp + (size_t)(j0 + jj * 16 + l15) * DHEAD;
      v16bf bk0 = *(const v16bf*)(krow + 16 * half);
      v16bf bk1 = *(const v16bf*)(krow + 32 + 16 * half);
      v8f sc = {};
      sc = wmma_bf16(aq0.v, bk0, sc);
      sc = wmma_bf16(aq1.v, bk1, sc);
      int kid = ids[b * LL + j0 + jj * 16 + l15];
#pragma unroll
      for (int v = 0; v < 8; v++)
        s[jj][v] = (qid[v] == kid) ? sc[v] : -3.402823466e38f;
    }
    float bm[8];
#pragma unroll
    for (int v = 0; v < 8; v++) bm[v] = fmaxf(s[0][v], s[1][v]);
    for (int d = 1; d < 16; d <<= 1)
#pragma unroll
      for (int v = 0; v < 8; v++) bm[v] = fmaxf(bm[v], __shfl_xor(bm[v], d, 32));
    float scale[8], psum[8];
#pragma unroll
    for (int v = 0; v < 8; v++) {
      float mn = fmaxf(mstat[v], bm[v]);
      scale[v] = __expf(mstat[v] - mn);
      mstat[v] = mn;
      psum[v] = 0.f;
    }
#pragma unroll
    for (int jj = 0; jj < 2; jj++)
#pragma unroll
      for (int v = 0; v < 8; v++) {
        float p = __expf(s[jj][v] - mstat[v]);
        psum[v] += p;
        P[wv][8 * half + v][jj * 16 + l15] = (bf16_t)p;
      }
    for (int d = 1; d < 16; d <<= 1)
#pragma unroll
      for (int v = 0; v < 8; v++) psum[v] += __shfl_xor(psum[v], d, 32);
#pragma unroll
    for (int v = 0; v < 8; v++) lstat[v] = lstat[v] * scale[v] + psum[v];
#pragma unroll
    for (int t = 0; t < 4; t++)
#pragma unroll
      for (int v = 0; v < 8; v++) o[t][v] *= scale[v];
    V16u ap;
    ap.h[0] = *(const bfx8*)&P[wv][l15][16 * half];
    ap.h[1] = *(const bfx8*)&P[wv][l15][8 + 16 * half];
#pragma unroll
    for (int t = 0; t < 4; t++) {
      v16bf bv = *(const v16bf*)(Vp + (size_t)(t * 16 + l15) * LL + j0 + 16 * half);
      o[t] = wmma_bf16(ap.v, bv, o[t]);
    }
  }
#pragma unroll
  for (int t = 0; t < 4; t++)
#pragma unroll
    for (int v = 0; v < 8; v++) {
      float x = o[t][v] / lstat[v];
      int q = q16 * 16 + 8 * half + v;
      AOb[((size_t)b * LL + q) * DMODEL + h * DHEAD + t * 16 + l15] = (bf16_t)x;
    }
}

// ---------------------------------------------------------------------------
// Pool query: qn = RMS( LN(pool_q)*g @ Wq ) * 8 * q_g  (512 values, shared by all)
// ---------------------------------------------------------------------------
__global__ void __launch_bounds__(256) pool_q_kernel(
    const float* __restrict__ pq, const float* __restrict__ lng,
    const float* __restrict__ Wq, const float* __restrict__ qg,
    float* __restrict__ qn)
{
  __shared__ float xn[DMODEL], qsh[DMODEL], sc[8];
  __shared__ float sh1[8], sh2[8], tot[2];
  int t = threadIdx.x;
  float v0 = pq[t], v1 = pq[t + 256];
  float s = v0 + v1, ss = v0 * v0 + v1 * v1;
  for (int d = 1; d < 32; d <<= 1) { s += __shfl_xor(s, d, 32); ss += __shfl_xor(ss, d, 32); }
  int wv = t >> 5, lane = t & 31;
  if (lane == 0) { sh1[wv] = s; sh2[wv] = ss; }
  __syncthreads();
  if (t == 0) {
    float a = 0.f, b = 0.f;
    for (int i = 0; i < 8; i++) { a += sh1[i]; b += sh2[i]; }
    tot[0] = a; tot[1] = b;
  }
  __syncthreads();
  float mean = tot[0] * (1.0f / DMODEL);
  float var  = tot[1] * (1.0f / DMODEL) - mean * mean;
  float rstd = rsqrtf(var + 1e-5f);
  xn[t]       = (v0 - mean) * rstd * lng[t];
  xn[t + 256] = (v1 - mean) * rstd * lng[t + 256];
  __syncthreads();
  for (int o = t; o < DMODEL; o += 256) {
    float acc = 0.f;
    for (int k = 0; k < DMODEL; k++) acc += xn[k] * Wq[(size_t)k * DMODEL + o];
    qsh[o] = acc;
  }
  __syncthreads();
  if (t < NH) {
    float a = 0.f;
    for (int d = 0; d < DHEAD; d++) { float x = qsh[t * DHEAD + d]; a += x * x; }
    sc[t] = 8.0f / fmaxf(sqrtf(a), 1e-12f);
  }
  __syncthreads();
  for (int o = t; o < DMODEL; o += 256)
    qn[o] = qsh[o] * sc[o >> 6] * qg[o];
}

// ---------------------------------------------------------------------------
// Pool attention: one wave per (b, img, h); iterates all L keys, fp32 online softmax.
// kv: [B, L, 1024] = [k(512) | v(512)], k gets RMS-norm on the fly.
// ---------------------------------------------------------------------------
__global__ void __launch_bounds__(32) pool_attn_kernel(
    const float* __restrict__ qn, const float* __restrict__ kv,
    const float* __restrict__ kg, const int* __restrict__ ids,
    float* __restrict__ PA)
{
  int id = blockIdx.x;
  int h = id & (NH - 1);
  int img = (id >> 3) & (NIMG_ - 1);
  int b = id >> 5;
  int lane = threadIdx.x;
  int d0 = lane * 2;
  float q0 = qn[h * DHEAD + d0], q1 = qn[h * DHEAD + d0 + 1];
  float g0 = kg[h * DHEAD + d0], g1 = kg[h * DHEAD + d0 + 1];
  float m = -__builtin_inff(), l = 0.f, a0 = 0.f, a1 = 0.f;
  for (int j = 0; j < LL; j++) {
    const float* row = kv + ((size_t)b * LL + j) * (2 * DMODEL);
    float k0 = row[h * DHEAD + d0], k1 = row[h * DHEAD + d0 + 1];
    float pd = q0 * k0 * g0 + q1 * k1 * g1;
    float pn = k0 * k0 + k1 * k1;
    for (int d = 1; d < 32; d <<= 1) { pd += __shfl_xor(pd, d, 32); pn += __shfl_xor(pn, d, 32); }
    float s;
    if (ids[b * LL + j] == img) s = pd * 8.0f / fmaxf(sqrtf(pn), 1e-12f);
    else s = -3.402823466e38f;
    float mn = fmaxf(m, s);
    float scale = __expf(m - mn);
    float p = __expf(s - mn);
    m = mn;
    l = l * scale + p;
    float v0 = row[DMODEL + h * DHEAD + d0], v1 = row[DMODEL + h * DHEAD + d0 + 1];
    a0 = a0 * scale + p * v0;
    a1 = a1 * scale + p * v1;
  }
  size_t ob = ((size_t)b * NIMG_ + img) * DMODEL + h * DHEAD + d0;
  PA[ob] = a0 / l;
  PA[ob + 1] = a1 / l;
}

// ---------------------------------------------------------------------------
// Small fp32 matmul: out[M,N] = A[M,K] @ W[K,N] (+ per-column addvec)
// ---------------------------------------------------------------------------
__global__ void small_mm_kernel(const float* __restrict__ A, const float* __restrict__ W,
                                const float* __restrict__ addv, float* __restrict__ outp,
                                int M, int N, int K) {
  int idx = blockIdx.x * 256 + threadIdx.x;
  if (idx >= M * N) return;
  int m = idx / N, n = idx - m * N;
  float acc = 0.f;
  for (int k = 0; k < K; k++) acc += A[(size_t)m * K + k] * W[(size_t)k * N + n];
  if (addv) acc += addv[n];
  outp[idx] = acc;
}

// ---------------------------------------------------------------------------
extern "C" void kernel_launch(void* const* d_in, const int* in_sizes, int n_in,
                              void* d_out, int out_size, void* d_ws, size_t ws_size,
                              hipStream_t stream) {
  (void)in_sizes; (void)n_in; (void)out_size; (void)ws_size;
  const float* patches    = (const float*)d_in[0];
  const int*   pos_h      = (const int*)d_in[1];
  const int*   pos_w      = (const int*)d_in[2];
  const int*   image_ids  = (const int*)d_in[3];
  const float* ln_pe1_g   = (const float*)d_in[4];
  const float* W_pe       = (const float*)d_in[5];
  const float* b_pe       = (const float*)d_in[6];
  const float* ln_pe2_g   = (const float*)d_in[7];
  const float* pos_h_emb  = (const float*)d_in[8];
  const float* pos_w_emb  = (const float*)d_in[9];
  const float* attn_ln_g  = (const float*)d_in[10];
  const float* q_g        = (const float*)d_in[11];
  const float* k_g        = (const float*)d_in[12];
  const float* Wq         = (const float*)d_in[13];
  const float* Wkv        = (const float*)d_in[14];
  const float* Wo         = (const float*)d_in[15];
  const float* ff_ln_g    = (const float*)d_in[16];
  const float* W1         = (const float*)d_in[17];
  const float* b1         = (const float*)d_in[18];
  const float* W2         = (const float*)d_in[19];
  const float* b2         = (const float*)d_in[20];
  const float* final_ln_g = (const float*)d_in[21];
  const float* pool_q     = (const float*)d_in[22];
  const float* pool_ln_g  = (const float*)d_in[23];
  const float* pool_q_g   = (const float*)d_in[24];
  const float* pool_k_g   = (const float*)d_in[25];
  const float* pool_Wq    = (const float*)d_in[26];
  const float* pool_Wkv   = (const float*)d_in[27];
  const float* pool_Wo    = (const float*)d_in[28];
  const float* head_ln_g  = (const float*)d_in[29];
  const float* W_head     = (const float*)d_in[30];
  float* outp = (float*)d_out;

  char* w = (char*)d_ws;
  size_t off = 0;
  auto alloc = [&](size_t bytes) { size_t o = off; off += (bytes + 255) & ~(size_t)255; return o; };
  const size_t ML = (size_t)BB * LL;   // 4096 rows

  size_t o_X   = alloc(ML * DMODEL * 4);
  size_t o_XB  = alloc(ML * DMODEL * 2);      // also attention output bf16
  size_t o_Q   = alloc(ML * DMODEL * 4);      // Qf; also patch-embed GEMM out
  size_t o_KV  = alloc(ML * 2 * DMODEL * 4);  // KVf; also PB bf16, H1B bf16
  size_t o_QB  = alloc(ML * DMODEL * 2);
  size_t o_KB  = alloc(ML * DMODEL * 2);
  size_t o_VT  = alloc(ML * DMODEL * 2);
  size_t o_QN  = alloc(DMODEL * 4);
  size_t o_PA  = alloc((size_t)BB * NIMG_ * DMODEL * 4);
  size_t o_PO  = alloc((size_t)BB * NIMG_ * DMODEL * 4);
  size_t o_PN  = alloc((size_t)BB * NIMG_ * DMODEL * 4);
  size_t o_WpeT = alloc((size_t)PDIM * DMODEL * 2);
  size_t o_WqT  = alloc((size_t)NDEPTH * DMODEL * DMODEL * 2);
  size_t o_WkvT = alloc((size_t)NDEPTH * DMODEL * 2 * DMODEL * 2);
  size_t o_WoT  = alloc((size_t)NDEPTH * DMODEL * DMODEL * 2);
  size_t o_W1T  = alloc((size_t)NDEPTH * DMODEL * DMLP * 2);
  size_t o_W2T  = alloc((size_t)NDEPTH * DMLP * DMODEL * 2);
  size_t o_PkvT = alloc((size_t)DMODEL * 2 * DMODEL * 2);

  float*  X    = (float*)(w + o_X);
  bf16_t* XB   = (bf16_t*)(w + o_XB);
  float*  Qf   = (float*)(w + o_Q);
  float*  TMPF = (float*)(w + o_Q);
  float*  KVf  = (float*)(w + o_KV);
  bf16_t* PB   = (bf16_t*)(w + o_KV);
  bf16_t* H1B  = (bf16_t*)(w + o_KV);
  bf16_t* Qb   = (bf16_t*)(w + o_QB);
  bf16_t* Kb   = (bf16_t*)(w + o_KB);
  bf16_t* Vt   = (bf16_t*)(w + o_VT);
  float*  QN   = (float*)(w + o_QN);
  float*  PA   = (float*)(w + o_PA);
  float*  PO   = (float*)(w + o_PO);
  float*  PN   = (float*)(w + o_PN);
  bf16_t* WpeT = (bf16_t*)(w + o_WpeT);
  bf16_t* WqT  = (bf16_t*)(w + o_WqT);
  bf16_t* WkvT = (bf16_t*)(w + o_WkvT);
  bf16_t* WoT  = (bf16_t*)(w + o_WoT);
  bf16_t* W1T  = (bf16_t*)(w + o_W1T);
  bf16_t* W2T  = (bf16_t*)(w + o_W2T);
  bf16_t* PkvT = (bf16_t*)(w + o_PkvT);

  auto convT = [&](const float* src, bf16_t* dst, int K, int N) {
    int total = K * N;
    convT_kernel<<<(total + 255) / 256, 256, 0, stream>>>(src, dst, K, N);
  };
  auto gemm = [&](const bf16_t* A, const bf16_t* Bt, const float* bias, const float* res,
                  float* oF, bf16_t* oB, int M, int N, int K, int gelu) {
    dim3 grid(N / 64, M / 64);
    gemm_bf16_kernel<<<grid, 128, 0, stream>>>(A, Bt, bias, res, oF, oB, M, N, K, gelu);
  };

  // ---- weight conversion (transposed bf16) ----
  convT(W_pe, WpeT, PDIM, DMODEL);
  for (int i = 0; i < NDEPTH; i++) {
    convT(Wq  + (size_t)i * DMODEL * DMODEL,     WqT  + (size_t)i * DMODEL * DMODEL,     DMODEL, DMODEL);
    convT(Wkv + (size_t)i * DMODEL * 2 * DMODEL, WkvT + (size_t)i * DMODEL * 2 * DMODEL, DMODEL, 2 * DMODEL);
    convT(Wo  + (size_t)i * DMODEL * DMODEL,     WoT  + (size_t)i * DMODEL * DMODEL,     DMODEL, DMODEL);
    convT(W1  + (size_t)i * DMODEL * DMLP,       W1T  + (size_t)i * DMODEL * DMLP,       DMODEL, DMLP);
    convT(W2  + (size_t)i * DMLP * DMODEL,       W2T  + (size_t)i * DMLP * DMODEL,       DMLP, DMODEL);
  }
  convT(pool_Wkv, PkvT, DMODEL, 2 * DMODEL);

  // ---- patch embedding ----
  ln_kernel<<<(unsigned)ML, 256, 0, stream>>>(patches, ln_pe1_g, nullptr, PB, PDIM);
  gemm(PB, WpeT, b_pe, nullptr, TMPF, nullptr, (int)ML, DMODEL, PDIM, 0);
  ln_pos_kernel<<<(unsigned)ML, 256, 0, stream>>>(TMPF, ln_pe2_g, pos_h_emb, pos_w_emb,
                                                  pos_h, pos_w, X);

  // ---- transformer layers ----
  for (int i = 0; i < NDEPTH; i++) {
    bf16_t* WqT_i  = WqT  + (size_t)i * DMODEL * DMODEL;
    bf16_t* WkvT_i = WkvT + (size_t)i * DMODEL * 2 * DMODEL;
    bf16_t* WoT_i  = WoT  + (size_t)i * DMODEL * DMODEL;
    bf16_t* W1T_i  = W1T  + (size_t)i * DMODEL * DMLP;
    bf16_t* W2T_i  = W2T  + (size_t)i * DMLP * DMODEL;
    ln_kernel<<<(unsigned)ML, 256, 0, stream>>>(X, attn_ln_g + i * DMODEL, nullptr, XB, DMODEL);
    gemm(XB, WqT_i,  nullptr, nullptr, Qf,  nullptr, (int)ML, DMODEL,     DMODEL, 0);
    gemm(XB, WkvT_i, nullptr, nullptr, KVf, nullptr, (int)ML, 2 * DMODEL, DMODEL, 0);
    rms_prep_kernel<<<(unsigned)ML, 256, 0, stream>>>(Qf, KVf,
        q_g + (size_t)i * NH * DHEAD, k_g + (size_t)i * NH * DHEAD, Qb, Kb, Vt);
    flash_attn_kernel<<<BB * NH * (LL / 16) / 4, 128, 0, stream>>>(Qb, Kb, Vt, image_ids, XB);
    gemm(XB, WoT_i, nullptr, X, X, nullptr, (int)ML, DMODEL, DMODEL, 0);
    ln_kernel<<<(unsigned)ML, 256, 0, stream>>>(X, ff_ln_g + i * DMODEL, nullptr, XB, DMODEL);
    gemm(XB, W1T_i, b1 + (size_t)i * DMLP, nullptr, nullptr, H1B, (int)ML, DMLP, DMODEL, 1);
    gemm(H1B, W2T_i, b2 + (size_t)i * DMODEL, X, X, nullptr, (int)ML, DMODEL, DMLP, 0);
  }

  // ---- final LN + attention pooling + head ----
  ln_kernel<<<(unsigned)ML, 256, 0, stream>>>(X, final_ln_g, nullptr, XB, DMODEL);
  gemm(XB, PkvT, nullptr, nullptr, KVf, nullptr, (int)ML, 2 * DMODEL, DMODEL, 0);
  pool_q_kernel<<<1, 256, 0, stream>>>(pool_q, pool_ln_g, pool_Wq, pool_q_g, QN);
  pool_attn_kernel<<<BB * NIMG_ * NH, 32, 0, stream>>>(QN, KVf, pool_k_g, image_ids, PA);
  small_mm_kernel<<<(BB * NIMG_ * DMODEL + 255) / 256, 256, 0, stream>>>(
      PA, pool_Wo, pool_q, PO, BB * NIMG_, DMODEL, DMODEL);
  ln_kernel<<<BB * NIMG_, 256, 0, stream>>>(PO, head_ln_g, PN, nullptr, DMODEL);
  small_mm_kernel<<<(BB * NIMG_ * NCLS_ + 255) / 256, 256, 0, stream>>>(
      PN, W_head, nullptr, outp, BB * NIMG_, NCLS_, DMODEL);
}
